// LayerNormLSTM_29240137351300
// MI455X (gfx1250) — matrix operand order repaired and checked by reading, not compiled, not run on hardware
//
#include <hip/hip_runtime.h>
#include <hip/hip_bf16.h>
#include <math.h>

// Problem constants (match reference)
#define TT 512
#define BB 64
#define DD 1024
#define HH 1024
#define GG 4096          // 4*H
#define LN_EPS 1e-5f

typedef __bf16 bf16_t;
typedef bf16_t v16bf __attribute__((ext_vector_type(16)));
typedef bf16_t v8bf  __attribute__((ext_vector_type(8)));
typedef float  v8f   __attribute__((ext_vector_type(8)));

// ---------------------------------------------------------------------------
// fp32 [B,T,D] -> bf16 [T,B,D] (time-major for the batched ih GEMM)
// ---------------------------------------------------------------------------
__global__ void convert_transpose_x(const float* __restrict__ x, bf16_t* __restrict__ xbf) {
  size_t idx = (size_t)blockIdx.x * blockDim.x + threadIdx.x;
  size_t total = (size_t)BB * TT * DD;
  if (idx >= total) return;
  size_t d  = idx % DD;
  size_t bt = idx / DD;
  size_t t  = bt % TT;
  size_t b  = bt / TT;
  xbf[((size_t)t * BB + b) * DD + d] = (bf16_t)x[idx];
}

__global__ void f32_to_bf16(const float* __restrict__ src, bf16_t* __restrict__ dst, size_t n) {
  size_t i = (size_t)blockIdx.x * blockDim.x + threadIdx.x;
  if (i < n) dst[i] = (bf16_t)src[i];
}

// ---------------------------------------------------------------------------
// Generic bf16 WMMA GEMM:  C[m,n] = sum_k A[m,k] * W[n,k] + bias[n]
//   A: [M,K] bf16 row-major, W: [N,K] bf16 row-major, C: [M,N] fp32.
// One wave computes a 32x64 tile: 2 A-frags x 4 B-frags -> 8 WMMAs / K-step.
// VGPR layouts (A/B: contiguous k order chosen by this code; C/D per CDNA5 ISA 7.12.2):
//   A 16x32 bf16 : lane L (m=L%16) holds K = k0 + (L/16)*16 + 0..15
//   B 32x16 bf16 : lane L (n=L%16) holds K = k0 + (L/16)*16 + 0..15
//   C/D 16x16 f32: lane L, vgpr r -> (m = (L/16)*8 + r, n = L%16)
// M%32==0, N%64==0, K%32==0 required (holds: 32768/64, 4096, 1024).
// ---------------------------------------------------------------------------
__global__ __launch_bounds__(128)
void gemm_bf16_wmma(const bf16_t* __restrict__ A, const bf16_t* __restrict__ W,
                    const float* __restrict__ bias, float* __restrict__ C,
                    int M, int N, int K) {
  const int lane = threadIdx.x & 31;
  const int wave = blockIdx.x * (blockDim.x >> 5) + (threadIdx.x >> 5);
  const int ntilesN = N >> 6;
  const int tm = wave / ntilesN;
  const int tn = wave % ntilesN;
  if (tm * 32 >= M) return;           // wave-uniform: EXEC stays all-ones for WMMA

  const int half = lane >> 4;         // K-half selector
  const int sub  = lane & 15;         // m (for A) / n (for B) within tile

  const bf16_t* a0row = A + (size_t)(tm * 32 + sub) * K;
  const bf16_t* a1row = a0row + (size_t)16 * K;
  const bf16_t* brow  = W + (size_t)(tn * 64 + sub) * K;

  const v8f vzero = {0.f, 0.f, 0.f, 0.f, 0.f, 0.f, 0.f, 0.f};
  v8f acc0[4], acc1[4];
#pragma unroll
  for (int j = 0; j < 4; ++j) { acc0[j] = vzero; acc1[j] = vzero; }

  for (int k0 = 0; k0 < K; k0 += 32) {
    union { v16bf v; v8bf h[2]; } a0u, a1u;
    a0u.h[0] = *(const v8bf*)(a0row + k0 + half * 16);
    a0u.h[1] = *(const v8bf*)(a0row + k0 + half * 16 + 8);
    a1u.h[0] = *(const v8bf*)(a1row + k0 + half * 16);
    a1u.h[1] = *(const v8bf*)(a1row + k0 + half * 16 + 8);
#pragma unroll
    for (int j = 0; j < 4; ++j) {
      v16bf bv = *(const v16bf*)(brow + (size_t)j * 16 * K + k0 + half * 16);
      acc0[j] = __builtin_amdgcn_wmma_f32_16x16x32_bf16(
          false, a0u.v, false, bv, (short)0, acc0[j], false, false);
      acc1[j] = __builtin_amdgcn_wmma_f32_16x16x32_bf16(
          false, a1u.v, false, bv, (short)0, acc1[j], false, false);
    }
  }

#pragma unroll
  for (int j = 0; j < 4; ++j) {
    int n = tn * 64 + j * 16 + sub;
    float bv = bias[n];
#pragma unroll
    for (int r = 0; r < 8; ++r) {
      int m0 = tm * 32 + half * 8 + r;
      C[(size_t)m0 * N + n]        = acc0[j][r] + bv;
      C[(size_t)(m0 + 16) * N + n] = acc1[j][r] + bv;
    }
  }
}

// ---------------------------------------------------------------------------
// In-place LayerNorm over rows of length ncols (one 256-thread block per row)
// ---------------------------------------------------------------------------
__global__ __launch_bounds__(256)
void ln_rows_kernel(float* __restrict__ data, const float* __restrict__ g,
                    const float* __restrict__ be, int ncols) {
  float* p = data + (size_t)blockIdx.x * ncols;
  int tid = threadIdx.x;
  float s = 0.f, s2 = 0.f;
  for (int i = tid; i < ncols; i += 256) { float v = p[i]; s += v; s2 += v * v; }
  __shared__ float sh[512];
  sh[tid] = s; sh[256 + tid] = s2;
  __syncthreads();
  for (int w = 128; w > 0; w >>= 1) {
    if (tid < w) { sh[tid] += sh[tid + w]; sh[256 + tid] += sh[256 + tid + w]; }
    __syncthreads();
  }
  float mu   = sh[0] / (float)ncols;
  float var  = sh[256] / (float)ncols - mu * mu;
  float rstd = rsqrtf(var + LN_EPS);
  for (int i = tid; i < ncols; i += 256)
    p[i] = (p[i] - mu) * rstd * g[i] + be[i];
}

__device__ __forceinline__ float sigm(float x) { return 1.0f / (1.0f + expf(-x)); }

// ---------------------------------------------------------------------------
// Fused: LN over hh preactivation row (4096) + gate combine + LSTM cell.
// One block per batch row. Writes h (fp32 to output slice, bf16 for next GEMM),
// updates c, optionally writes the "last timestep" tail output.
// ---------------------------------------------------------------------------
__global__ __launch_bounds__(256)
void lstm_cell_kernel(const float* __restrict__ tmp,   // [B,G] h@Whh^T + b_hh (pre-LN)
                      const float* __restrict__ gih,   // [B,G] LN'd ih gates at time t
                      const float* __restrict__ ghh, const float* __restrict__ behh,
                      float* __restrict__ c,           // [B,H] state (fp32)
                      bf16_t* __restrict__ hbf,        // [B,H] next-step GEMM input
                      float* __restrict__ out_t,       // d_out + t*2H + dir*H (b-stride T*2H)
                      float* __restrict__ out_last) {  // nullable: tail + dir*H (b-stride 2H)
  int b = blockIdx.x;
  int tid = threadIdx.x;
  const float* p = tmp + (size_t)b * GG;

  float s = 0.f, s2 = 0.f;
  for (int i = tid; i < GG; i += 256) { float v = p[i]; s += v; s2 += v * v; }
  __shared__ float sh[512];
  sh[tid] = s; sh[256 + tid] = s2;
  __syncthreads();
  for (int w = 128; w > 0; w >>= 1) {
    if (tid < w) { sh[tid] += sh[tid + w]; sh[256 + tid] += sh[256 + tid + w]; }
    __syncthreads();
  }
  float mu   = sh[0] / (float)GG;
  float rstd = rsqrtf(sh[256] / (float)GG - mu * mu + LN_EPS);

  const float* gi_row = gih + (size_t)b * GG;
  for (int i = tid; i < HH; i += 256) {
    float xi = gi_row[i]          + (p[i]          - mu) * rstd * ghh[i]          + behh[i];
    float xf = gi_row[HH + i]     + (p[HH + i]     - mu) * rstd * ghh[HH + i]     + behh[HH + i];
    float xg = gi_row[2 * HH + i] + (p[2 * HH + i] - mu) * rstd * ghh[2 * HH + i] + behh[2 * HH + i];
    float xo = gi_row[3 * HH + i] + (p[3 * HH + i] - mu) * rstd * ghh[3 * HH + i] + behh[3 * HH + i];
    size_t si = (size_t)b * HH + i;
    float cp = c[si];
    float cn = sigm(xf) * cp + sigm(xi) * tanhf(xg);
    float hn = sigm(xo) * tanhf(cn);
    c[si] = cn;
    hbf[si] = (bf16_t)hn;
    out_t[(size_t)b * ((size_t)TT * 2 * HH) + i] = hn;
    if (out_last) out_last[(size_t)b * (2 * HH) + i] = hn;
  }
}

// init h (bf16) and c (fp32) from a [2,H] init tensor, broadcast over batch
__global__ void init_state_kernel(const float* __restrict__ init2,
                                  bf16_t* __restrict__ hbf, float* __restrict__ c) {
  int idx = blockIdx.x * blockDim.x + threadIdx.x;
  if (idx >= BB * HH) return;
  int h = idx % HH;
  hbf[idx] = (bf16_t)init2[h];
  c[idx]   = init2[HH + h];
}

// ---------------------------------------------------------------------------
extern "C" void kernel_launch(void* const* d_in, const int* in_sizes, int n_in,
                              void* d_out, int out_size, void* d_ws, size_t ws_size,
                              hipStream_t stream) {
  const float* x       = (const float*)d_in[0];
  const float* w_ih_f  = (const float*)d_in[1];
  const float* w_hh_f  = (const float*)d_in[2];
  const float* b_ih_f  = (const float*)d_in[3];
  const float* b_hh_f  = (const float*)d_in[4];
  const float* g_ih_f  = (const float*)d_in[5];
  const float* be_ih_f = (const float*)d_in[6];
  const float* g_hh_f  = (const float*)d_in[7];
  const float* be_hh_f = (const float*)d_in[8];
  const float* w_ih_b  = (const float*)d_in[9];
  const float* w_hh_b  = (const float*)d_in[10];
  const float* b_ih_b  = (const float*)d_in[11];
  const float* b_hh_b  = (const float*)d_in[12];
  const float* g_ih_b  = (const float*)d_in[13];
  const float* be_ih_b = (const float*)d_in[14];
  const float* g_hh_b  = (const float*)d_in[15];
  const float* be_hh_b = (const float*)d_in[16];
  const float* fwd_init = (const float*)d_in[17];
  const float* bwd_init = (const float*)d_in[18];
  float* out = (float*)d_out;
  (void)in_sizes; (void)n_in; (void)out_size; (void)ws_size;

  // Workspace layout (~1.17 GB)
  char* ws = (char*)d_ws;
  size_t off = 0;
  auto take = [&](size_t bytes) -> char* {
    char* p = ws + off;
    off = (off + bytes + 255) & ~(size_t)255;
    return p;
  };
  bf16_t* xbf    = (bf16_t*)take((size_t)TT * BB * DD * 2);
  bf16_t* wihf   = (bf16_t*)take((size_t)GG * DD * 2);
  bf16_t* whhf   = (bf16_t*)take((size_t)GG * HH * 2);
  bf16_t* wihb   = (bf16_t*)take((size_t)GG * DD * 2);
  bf16_t* whhb   = (bf16_t*)take((size_t)GG * HH * 2);
  float*  gatesF = (float*)take((size_t)TT * BB * GG * 4);
  float*  gatesB = (float*)take((size_t)TT * BB * GG * 4);
  float*  tmpF   = (float*)take((size_t)BB * GG * 4);
  float*  tmpB   = (float*)take((size_t)BB * GG * 4);
  bf16_t* hbfF   = (bf16_t*)take((size_t)BB * HH * 2);
  bf16_t* hbfB   = (bf16_t*)take((size_t)BB * HH * 2);
  float*  cF     = (float*)take((size_t)BB * HH * 4);
  float*  cB     = (float*)take((size_t)BB * HH * 4);

  // 1) convert inputs to bf16 (x also transposed to time-major)
  {
    size_t n = (size_t)BB * TT * DD;
    convert_transpose_x<<<(unsigned)((n + 255) / 256), 256, 0, stream>>>(x, xbf);
  }
  auto cvt = [&](const float* s, bf16_t* d, size_t n) {
    f32_to_bf16<<<(unsigned)((n + 255) / 256), 256, 0, stream>>>(s, d, n);
  };
  cvt(w_ih_f, wihf, (size_t)GG * DD);
  cvt(w_hh_f, whhf, (size_t)GG * HH);
  cvt(w_ih_b, wihb, (size_t)GG * DD);
  cvt(w_hh_b, whhb, (size_t)GG * HH);

  // 2) batched input-side gates: LN(x W_ih^T + b_ih) for both directions
  {
    int M = TT * BB, N = GG, K = DD;
    int waves  = (M / 32) * (N / 64);
    int blocks = waves / 4;  // 128 threads = 4 waves per block
    gemm_bf16_wmma<<<blocks, 128, 0, stream>>>(xbf, wihf, b_ih_f, gatesF, M, N, K);
    ln_rows_kernel<<<M, 256, 0, stream>>>(gatesF, g_ih_f, be_ih_f, N);
    gemm_bf16_wmma<<<blocks, 128, 0, stream>>>(xbf, wihb, b_ih_b, gatesB, M, N, K);
    ln_rows_kernel<<<M, 256, 0, stream>>>(gatesB, g_ih_b, be_ih_b, N);
  }

  // 3) initial states
  {
    int n = BB * HH;
    init_state_kernel<<<(n + 255) / 256, 256, 0, stream>>>(fwd_init, hbfF, cF);
    init_state_kernel<<<(n + 255) / 256, 256, 0, stream>>>(bwd_init, hbfB, cB);
  }

  // 4) recurrence: 512 steps x {hh-GEMM, fused LN+cell} per direction.
  //    bwd consumes x[T-1-t] (reversed input) but writes output position t
  //    (reference does NOT flip bwd outputs back).
  {
    int Ms = BB, N = GG, K = HH;
    int wavesS  = (Ms / 32) * (N / 64);  // 128 waves
    int blocksS = wavesS / 4;            // 32 blocks
    float* tail = out + (size_t)BB * TT * 2 * HH;  // final[:, -1, :]
    for (int t = 0; t < TT; ++t) {
      // forward direction
      gemm_bf16_wmma<<<blocksS, 128, 0, stream>>>(hbfF, whhf, b_hh_f, tmpF, Ms, N, K);
      lstm_cell_kernel<<<BB, 256, 0, stream>>>(
          tmpF, gatesF + (size_t)t * BB * GG, g_hh_f, be_hh_f, cF, hbfF,
          out + (size_t)t * (2 * HH),
          (t == TT - 1) ? tail : (float*)nullptr);
      // backward direction
      gemm_bf16_wmma<<<blocksS, 128, 0, stream>>>(hbfB, whhb, b_hh_b, tmpB, Ms, N, K);
      lstm_cell_kernel<<<BB, 256, 0, stream>>>(
          tmpB, gatesB + (size_t)(TT - 1 - t) * BB * GG, g_hh_b, be_hh_b, cB, hbfB,
          out + (size_t)t * (2 * HH) + HH,
          (t == TT - 1) ? (tail + HH) : (float*)nullptr);
    }
  }
}
